// GmapAD_GAT_6700148982128
// MI455X (gfx1250) — matrix operand order repaired and checked by reading, not compiled, not run on hardware
//
#include <hip/hip_runtime.h>

typedef __bf16 bf16_t;
typedef bf16_t v16bf __attribute__((ext_vector_type(16)));
typedef float  v8f   __attribute__((ext_vector_type(8)));

#define GAT_SLOPE 0.2f
#define ACT_SLOPE 0.01f

__device__ __forceinline__ float lrelu(float x, float s) { return x > 0.f ? x : x * s; }

// float atomic max via ordered-int trick; slot must be initialized to -inf.
__device__ __forceinline__ void atomic_max_f32(float* addr, float val) {
  if (val >= 0.f)
    atomicMax((int*)addr, __float_as_int(val));
  else
    atomicMin((unsigned int*)addr, __float_as_uint(val));
}

// ---------------------------------------------------------------------------
// C[M,Ncols] = A[M,K] @ B[K,Ncols]; fp32 in/out, bf16 WMMA compute, f32 accum.
// One wave computes a 16 x (16*CT) output strip: the A fragment is loaded and
// converted once per K-step and reused across CT column tiles (4x less A
// traffic / cvt work than one-tile-per-wave). K is a template constant so the
// K-loop fully unrolls. Fragment layouts (A/B in contiguous k order, not the ISA table order):
//   A (16x32 bf16): lanes 0-15 row M=l16, K {0..15}; lanes 16-31 K {16..31}
//   B (32x16 bf16): b[i] = B[k0 + 16*half + i][col]; col = l16
//   C/D (16x16 f32): VGPR r -> row r + 8*half, col l16
// ---------------------------------------------------------------------------
template <int K, int CT>
__global__ __launch_bounds__(128) void wmma_gemm_bf16(
    const float* __restrict__ A, const float* __restrict__ B,
    float* __restrict__ C, int M, int Ncols, int totTiles) {
  const int wave = threadIdx.x >> 5;
  const int lane = threadIdx.x & 31;
  const int tile = blockIdx.x * 4 + wave;
  if (tile >= totTiles) return;
  const int tilesN = Ncols / (16 * CT);
  const int row0 = (tile / tilesN) << 4;
  const int col0 = (tile % tilesN) * (16 * CT);
  const int half = lane >> 4;  // 0: lanes 0-15, 1: lanes 16-31
  const int l16  = lane & 15;

  v8f acc[CT] = {};
  const float* arow = A + (size_t)(row0 + l16) * K;

#pragma unroll
  for (int kk = 0; kk < K; kk += 32) {
    const float* ap = arow + kk + 16 * half;
    if (kk + 32 < K) __builtin_prefetch(ap + 32, 0, 0);
    v16bf a;
#pragma unroll
    for (int i = 0; i < 8; ++i) a[i] = (bf16_t)ap[i];
#pragma unroll
    for (int i = 0; i < 8; ++i) a[8 + i] = (bf16_t)ap[8 + i];

#pragma unroll
    for (int ct = 0; ct < CT; ++ct) {
      const float* bp = B + (size_t)(kk + 16 * half) * Ncols + col0 + ct * 16 + l16;
      v16bf b;
#pragma unroll
      for (int i = 0; i < 16; ++i) b[i] = (bf16_t)bp[(size_t)i * Ncols];
      acc[ct] = __builtin_amdgcn_wmma_f32_16x16x32_bf16(false, a, false, b,
                                                        (short)0, acc[ct], false, false);
    }
  }

#pragma unroll
  for (int ct = 0; ct < CT; ++ct)
#pragma unroll
    for (int r = 0; r < 8; ++r)
      C[(size_t)(row0 + r + 8 * half) * Ncols + col0 + ct * 16 + l16] = acc[ct][r];
}

// ---------------------------------------------------------------------------
// Per-(node, head) attention scalars: as = <xw[n,h,:], a_src[h,:]>, ad likewise
// ---------------------------------------------------------------------------
template <int H, int Ch>
__global__ __launch_bounds__(256) void alpha_kernel(
    const float* __restrict__ xw, const float* __restrict__ a_src,
    const float* __restrict__ a_dst, float* __restrict__ as,
    float* __restrict__ ad, int N) {
  int t = blockIdx.x * blockDim.x + threadIdx.x;
  if (t >= N * H) return;
  int n = t / H, h = t % H;
  const float* row = xw + (size_t)n * H * Ch + (size_t)h * Ch;
  float ssum = 0.f, dsum = 0.f;
#pragma unroll
  for (int c = 0; c < Ch; ++c) {
    float v = row[c];
    ssum += v * a_src[h * Ch + c];
    dsum += v * a_dst[h * Ch + c];
  }
  as[t] = ssum;
  ad[t] = dsum;
}

__global__ __launch_bounds__(256) void fill_kernel(float* __restrict__ p, float v, long long n) {
  long long t = (long long)blockIdx.x * blockDim.x + threadIdx.x;
  if (t < n) p[t] = v;
}

// edge passes: e index in [0, E+N); e >= E are self loops
__device__ __forceinline__ void edge_endpoints(const long long* ei, long long E,
                                               long long e, int& src, int& dst) {
  if (e < E) { src = (int)ei[e]; dst = (int)ei[E + e]; }
  else       { src = dst = (int)(e - E); }
}

template <int H>
__global__ __launch_bounds__(256) void edge_max_kernel(
    const long long* __restrict__ ei, long long E, int N,
    const float* __restrict__ as, const float* __restrict__ ad,
    float* __restrict__ m) {
  long long t = (long long)blockIdx.x * blockDim.x + threadIdx.x;
  long long total = (E + N) * H;
  if (t >= total) return;
  long long e = t / H;
  int h = (int)(t % H), src, dst;
  edge_endpoints(ei, E, e, src, dst);
  float v = lrelu(as[src * H + h] + ad[dst * H + h], GAT_SLOPE);
  atomic_max_f32(&m[dst * H + h], v);
}

template <int H>
__global__ __launch_bounds__(256) void edge_sum_kernel(
    const long long* __restrict__ ei, long long E, int N,
    const float* __restrict__ as, const float* __restrict__ ad,
    const float* __restrict__ m, float* __restrict__ den) {
  long long t = (long long)blockIdx.x * blockDim.x + threadIdx.x;
  long long total = (E + N) * H;
  if (t >= total) return;
  long long e = t / H;
  int h = (int)(t % H), src, dst;
  edge_endpoints(ei, E, e, src, dst);
  float v = lrelu(as[src * H + h] + ad[dst * H + h], GAT_SLOPE);
  atomicAdd(&den[dst * H + h], __expf(v - m[dst * H + h]));
}

// agg[dst, :] += alpha(e,h) * xw[src, :]; one thread handles 4 channels
template <int H, int Ch, int Cout>
__global__ __launch_bounds__(256) void edge_agg_kernel(
    const long long* __restrict__ ei, long long E, int N,
    const float* __restrict__ as, const float* __restrict__ ad,
    const float* __restrict__ m, const float* __restrict__ den,
    const float* __restrict__ xw, float* __restrict__ agg) {
  const int groups = Cout >> 2;
  long long t = (long long)blockIdx.x * blockDim.x + threadIdx.x;
  long long total = (E + N) * groups;
  if (t >= total) return;
  long long e = t / groups;
  int g = (int)(t % groups), src, dst;
  edge_endpoints(ei, E, e, src, dst);
  int c0 = g << 2;
  int h = c0 / Ch;
  float v = lrelu(as[src * H + h] + ad[dst * H + h], GAT_SLOPE);
  float alpha = __expf(v - m[dst * H + h]) / den[dst * H + h];
  const float4 xv = *(const float4*)(xw + (size_t)src * Cout + c0);
  float* o = agg + (size_t)dst * Cout + c0;
  atomicAdd(o + 0, alpha * xv.x);
  atomicAdd(o + 1, alpha * xv.y);
  atomicAdd(o + 2, alpha * xv.z);
  atomicAdd(o + 3, alpha * xv.w);
}

template <int C>
__global__ __launch_bounds__(256) void bias_lrelu_kernel(
    const float* __restrict__ agg, const float* __restrict__ b,
    float* __restrict__ out, long long n) {
  long long t = (long long)blockIdx.x * blockDim.x + threadIdx.x;
  if (t >= n) return;
  out[t] = lrelu(agg[t] + b[t & (C - 1)], ACT_SLOPE);
}

// partial column sums of n_reps[N,128] -> gacc[128]
__global__ __launch_bounds__(128) void grep_partial_kernel(
    const float* __restrict__ nreps, float* __restrict__ gacc, int N) {
  const int NPB = 256;
  int c = threadIdx.x;
  int n0 = blockIdx.x * NPB;
  int n1 = min(n0 + NPB, N);
  float s = 0.f;
  for (int n = n0; n < n1; ++n) s += nreps[(size_t)n * 128 + c];
  atomicAdd(&gacc[c], s);
}

__global__ __launch_bounds__(128) void final_kernel(
    const float* __restrict__ gacc, const float* __restrict__ fcw,
    const float* __restrict__ fcb, float* __restrict__ out_cls,
    float* __restrict__ out_grep, int N) {
  __shared__ float g[128];
  int c = threadIdx.x;
  float v = gacc[c] / (float)N;
  g[c] = v;
  out_grep[c] = v;
  __syncthreads();
  if (c == 0) {
    float l0 = fcb[0], l1 = fcb[1];
    for (int i = 0; i < 128; ++i) { l0 += g[i] * fcw[i * 2 + 0]; l1 += g[i] * fcw[i * 2 + 1]; }
    float mx = fmaxf(l0, l1);
    float e0 = __expf(l0 - mx), e1 = __expf(l1 - mx);
    out_cls[0] = e0 / (e0 + e1);
    out_cls[1] = e1 / (e0 + e1);
  }
}

static inline int blocks_for(long long total, int bs) { return (int)((total + bs - 1) / bs); }

extern "C" void kernel_launch(void* const* d_in, const int* in_sizes, int n_in,
                              void* d_out, int out_size, void* d_ws, size_t ws_size,
                              hipStream_t stream) {
  const float*     x      = (const float*)d_in[0];
  const long long* ei     = (const long long*)d_in[1];
  const float*     W1     = (const float*)d_in[2];
  const float*     a_src1 = (const float*)d_in[3];
  const float*     a_dst1 = (const float*)d_in[4];
  const float*     b1     = (const float*)d_in[5];
  const float*     W2     = (const float*)d_in[6];
  const float*     a_src2 = (const float*)d_in[7];
  const float*     a_dst2 = (const float*)d_in[8];
  const float*     b2     = (const float*)d_in[9];
  const float*     fc_w   = (const float*)d_in[10];
  const float*     fc_b   = (const float*)d_in[11];

  const int       F_in = 128, H = 8, Ch1 = 32, C1 = 256, Ch2 = 16, C2 = 128;
  const int       N = in_sizes[0] / F_in;
  const long long E = in_sizes[1] / 2;
  const long long EN = E + N;

  // workspace layout
  float* xw   = (float*)d_ws;                   // N*256 (layer2 reuses first N*128)
  float* hbuf = xw + (size_t)N * C1;            // N*256
  float* as   = hbuf + (size_t)N * C1;          // N*H
  float* ad   = as + (size_t)N * H;             // N*H
  float* mm   = ad + (size_t)N * H;             // N*H
  float* den  = mm + (size_t)N * H;             // N*H
  float* gacc = den + (size_t)N * H;            // 128

  float* out   = (float*)d_out;
  float* cls   = out;                           // [2]
  float* nreps = out + 2;                       // [N*128]
  float* grep  = out + 2 + (size_t)N * C2;      // [128]

  const int BS = 256;

  // ---------------- Layer 1 ----------------
  {
    int tiles = (N / 16) * (C1 / 64);  // 16x64 strips
    wmma_gemm_bf16<128, 4><<<blocks_for(tiles, 4), 128, 0, stream>>>(x, W1, xw, N, C1, tiles);
  }
  alpha_kernel<8, 32><<<blocks_for((long long)N * H, BS), BS, 0, stream>>>(xw, a_src1, a_dst1, as, ad, N);
  fill_kernel<<<blocks_for((long long)N * H, BS), BS, 0, stream>>>(mm, -__builtin_inff(), (long long)N * H);
  hipMemsetAsync(den, 0, (size_t)N * H * sizeof(float), stream);
  hipMemsetAsync(hbuf, 0, (size_t)N * C1 * sizeof(float), stream);
  edge_max_kernel<8><<<blocks_for(EN * H, BS), BS, 0, stream>>>(ei, E, N, as, ad, mm);
  edge_sum_kernel<8><<<blocks_for(EN * H, BS), BS, 0, stream>>>(ei, E, N, as, ad, mm, den);
  edge_agg_kernel<8, 32, 256><<<blocks_for(EN * (C1 / 4), BS), BS, 0, stream>>>(ei, E, N, as, ad, mm, den, xw, hbuf);
  bias_lrelu_kernel<256><<<blocks_for((long long)N * C1, BS), BS, 0, stream>>>(hbuf, b1, hbuf, (long long)N * C1);

  // ---------------- Layer 2 ----------------
  {
    int tiles = (N / 16) * (C2 / 64);
    wmma_gemm_bf16<256, 4><<<blocks_for(tiles, 4), 128, 0, stream>>>(hbuf, W2, xw, N, C2, tiles);
  }
  alpha_kernel<8, 16><<<blocks_for((long long)N * H, BS), BS, 0, stream>>>(xw, a_src2, a_dst2, as, ad, N);
  fill_kernel<<<blocks_for((long long)N * H, BS), BS, 0, stream>>>(mm, -__builtin_inff(), (long long)N * H);
  hipMemsetAsync(den, 0, (size_t)N * H * sizeof(float), stream);
  hipMemsetAsync(nreps, 0, (size_t)N * C2 * sizeof(float), stream);
  edge_max_kernel<8><<<blocks_for(EN * H, BS), BS, 0, stream>>>(ei, E, N, as, ad, mm);
  edge_sum_kernel<8><<<blocks_for(EN * H, BS), BS, 0, stream>>>(ei, E, N, as, ad, mm, den);
  edge_agg_kernel<8, 16, 128><<<blocks_for(EN * (C2 / 4), BS), BS, 0, stream>>>(ei, E, N, as, ad, mm, den, xw, nreps);
  bias_lrelu_kernel<128><<<blocks_for((long long)N * C2, BS), BS, 0, stream>>>(nreps, b2, nreps, (long long)N * C2);

  // ---------------- Readout ----------------
  hipMemsetAsync(gacc, 0, 128 * sizeof(float), stream);
  grep_partial_kernel<<<blocks_for(N, 256), 128, 0, stream>>>(nreps, gacc, N);
  final_kernel<<<1, 128, 0, stream>>>(gacc, fc_w, fc_b, cls, grep, N);
}
